// Net_37194416783911
// MI455X (gfx1250) — matrix operand order repaired and checked by reading, not compiled, not run on hardware
//
#include <hip/hip_runtime.h>
#include <cmath>

typedef __attribute__((ext_vector_type(16))) _Float16 v16h;
typedef __attribute__((ext_vector_type(8)))  float    v8f;
typedef __attribute__((ext_vector_type(8)))  _Float16 h8;

#define N_NODES 50000
#define N_EDGES 800000
#define DIM_IN  128
#define DIM_H   128
#define DIM_OUT 40
#define DIM_OUT_PAD 48

// ---------------- degree / norm ----------------
__global__ void fill_deg(float* __restrict__ deg, int n) {
    int i = blockIdx.x * blockDim.x + threadIdx.x;
    if (i < n) deg[i] = 1.0f;               // self-loop contributes 1
}

__global__ void count_deg(const long long* __restrict__ dst, float* __restrict__ deg, int e) {
    int i = blockIdx.x * blockDim.x + threadIdx.x;
    if (i < e) unsafeAtomicAdd(&deg[(int)dst[i]], 1.0f);
}

__global__ void rsqrt_deg(float* __restrict__ deg, int n) {
    int i = blockIdx.x * blockDim.x + threadIdx.x;
    if (i < n) deg[i] = rsqrtf(deg[i]);     // in-place -> dinv
}

// ---------------- fp32 -> fp16 converts ----------------
__global__ void convert_x_f16(const float* __restrict__ x, _Float16* __restrict__ xh, int total) {
    int i = blockIdx.x * blockDim.x + threadIdx.x;
    if (i < total) xh[i] = (_Float16)x[i];
}

// W is (K x Creal) row-major; Wt is (Cpad x K) row-major f16 (transposed, zero-padded cols)
__global__ void convert_w_t(const float* __restrict__ W, _Float16* __restrict__ Wt,
                            int K, int Creal, int Cpad) {
    int tid = blockIdx.x * blockDim.x + threadIdx.x;
    if (tid >= Cpad * K) return;
    int c = tid / K;
    int k = tid - c * K;
    Wt[tid] = (c < Creal) ? (_Float16)W[k * Creal + c] : (_Float16)0.0f;
}

// ---------------- WMMA GEMM: C[n x (16*nColTiles)] = A[n x K] * Bt^T ----------------
// A: f16 row-major, ld=K.  Bt: f16, transposed weights [col][K].  C: f32, ld=ldc.
__global__ __launch_bounds__(128)
void gemm_wmma_f16(const _Float16* __restrict__ A, const _Float16* __restrict__ Bt,
                   float* __restrict__ C, int nRowTiles, int nColTiles, int K, int ldc) {
    int wave = blockIdx.x * (blockDim.x >> 5) + (threadIdx.x >> 5);
    if (wave >= nRowTiles) return;
    int lane = threadIdx.x & 31;
    int r  = lane & 15;
    int hi = lane >> 4;
    int rowBase = wave * 16;

    const _Float16* arow = A + (size_t)(rowBase + r) * K;

    for (int ct = 0; ct < nColTiles; ++ct) {
        int col = ct * 16 + r;
        const _Float16* bcol = Bt + (size_t)col * K + hi * 8;

        v8f acc = {0.f, 0.f, 0.f, 0.f, 0.f, 0.f, 0.f, 0.f};
        #pragma unroll
        for (int k0 = 0; k0 < DIM_H; k0 += 32) {
            // A fragment: a[0..7]=A[r][k0+8hi .. +7], a[8..15]=A[r][k0+8hi+16 .. +23]
            h8 a0 = *(const h8*)(arow + k0 + hi * 8);
            h8 a1 = *(const h8*)(arow + k0 + hi * 8 + 16);
            // B fragment: b[0..7]=B[k0+8hi .. +7][col], b[8..15]=B[k0+8hi+16 .. +23][col]
            h8 b0 = *(const h8*)(bcol + k0);
            h8 b1 = *(const h8*)(bcol + k0 + 16);
            v16h a, b;
            #pragma unroll
            for (int i = 0; i < 8; ++i) { a[i] = a0[i]; a[i + 8] = a1[i]; }
            #pragma unroll
            for (int i = 0; i < 8; ++i) { b[i] = b0[i]; b[i + 8] = b1[i]; }
            acc = __builtin_amdgcn_wmma_f32_16x16x32_f16(
                false, a, false, b, (short)0, acc, false, false);
        }
        // D layout: VGPR j -> row rowBase + 8*hi + j, col = ct*16 + r
        float* crow = C + (size_t)(rowBase + hi * 8) * ldc + ct * 16 + r;
        #pragma unroll
        for (int j = 0; j < 8; ++j) crow[(size_t)j * ldc] = acc[j];
    }
}

// ---------------- self-loop + bias init:  agg = h * dinv^2 + b ----------------
__global__ void self_init(const float* __restrict__ h, const float* __restrict__ dinv,
                          const float* __restrict__ b, float* __restrict__ agg,
                          int n, int Cpad, int Creal) {
    int tid = blockIdx.x * blockDim.x + threadIdx.x;
    if (tid >= n * Cpad) return;
    int i = tid / Cpad;
    int c = tid - i * Cpad;
    float d = dinv[i];
    agg[tid] = h[tid] * d * d + (c < Creal ? b[c] : 0.0f);
}

// ---------------- edge scatter: agg[dst] += h[src] * dinv[src]*dinv[dst] ----------------
__global__ void edge_scatter(const long long* __restrict__ src, const long long* __restrict__ dst,
                             const float* __restrict__ h, const float* __restrict__ dinv,
                             float* __restrict__ agg, int e, int G, int ld) {
    int tid = blockIdx.x * blockDim.x + threadIdx.x;
    if (tid >= e * G) return;
    int ei = tid / G;
    int g  = tid - ei * G;
    int s = (int)src[ei];
    int d = (int)dst[ei];
    float w = dinv[s] * dinv[d];
    const float4 v = *(const float4*)(h + (size_t)s * ld + g * 4);
    float* o = agg + (size_t)d * ld + g * 4;
    unsafeAtomicAdd(o + 0, v.x * w);
    unsafeAtomicAdd(o + 1, v.y * w);
    unsafeAtomicAdd(o + 2, v.z * w);
    unsafeAtomicAdd(o + 3, v.w * w);
}

// ---------------- relu + convert to f16 for next layer ----------------
__global__ void relu_to_f16(const float* __restrict__ agg, _Float16* __restrict__ xh, int total) {
    int i = blockIdx.x * blockDim.x + threadIdx.x;
    if (i < total) xh[i] = (_Float16)fmaxf(agg[i], 0.0f);
}

// ---------------- relu + log_softmax over 40 cols (one wave32 per node) ----------------
__global__ __launch_bounds__(256)
void relu_logsoftmax40(const float* __restrict__ agg, float* __restrict__ out, int n, int ldin) {
    int node = blockIdx.x * (blockDim.x >> 5) + (threadIdx.x >> 5);
    if (node >= n) return;
    int lane = threadIdx.x & 31;
    const float* row = agg + (size_t)node * ldin;
    float v0 = fmaxf(row[lane], 0.0f);                                  // cols 0..31
    float v1 = (lane < 8) ? fmaxf(row[lane + 32], 0.0f) : -1e30f;       // cols 32..39
    float m = fmaxf(v0, v1);
    #pragma unroll
    for (int off = 16; off > 0; off >>= 1) m = fmaxf(m, __shfl_xor(m, off, 32));
    float s = expf(v0 - m) + ((lane < 8) ? expf(v1 - m) : 0.0f);
    #pragma unroll
    for (int off = 16; off > 0; off >>= 1) s += __shfl_xor(s, off, 32);
    float lz = m + logf(s);
    float* orow = out + (size_t)node * DIM_OUT;
    orow[lane] = v0 - lz;
    if (lane < 8) orow[lane + 32] = v1 - lz;
}

// ---------------- host side ----------------
static inline int cdiv(long long a, long long b) { return (int)((a + b - 1) / b); }

extern "C" void kernel_launch(void* const* d_in, const int* in_sizes, int n_in,
                              void* d_out, int out_size, void* d_ws, size_t ws_size,
                              hipStream_t stream) {
    const float*     x   = (const float*)d_in[0];
    const long long* ei  = (const long long*)d_in[1];   // (2, E) int64
    const float*     W0  = (const float*)d_in[2];
    const float*     b0  = (const float*)d_in[3];
    const float*     W1  = (const float*)d_in[4];
    const float*     b1  = (const float*)d_in[5];
    const float*     W2  = (const float*)d_in[6];
    const float*     b2  = (const float*)d_in[7];
    float*           out = (float*)d_out;

    const long long* src = ei;
    const long long* dst = ei + N_EDGES;

    // workspace carve-out (256B aligned)
    char* p = (char*)d_ws;
    size_t off = 0;
    auto carve = [&](size_t bytes) -> void* {
        void* q = p + off;
        off = (off + bytes + 255) & ~(size_t)255;
        return q;
    };
    _Float16* xh   = (_Float16*)carve((size_t)N_NODES * DIM_H * 2);
    float*    h    = (float*)   carve((size_t)N_NODES * DIM_H * 4);
    float*    agg  = (float*)   carve((size_t)N_NODES * DIM_H * 4);
    float*    dinv = (float*)   carve((size_t)N_NODES * 4);
    _Float16* Wt0  = (_Float16*)carve((size_t)DIM_H * DIM_H * 2);
    _Float16* Wt1  = (_Float16*)carve((size_t)DIM_H * DIM_H * 2);
    _Float16* Wt2  = (_Float16*)carve((size_t)DIM_OUT_PAD * DIM_H * 2);

    const int B = 256;
    const int nRowTiles = N_NODES / 16;          // 3125 exactly
    const int gemmBlocks = cdiv(nRowTiles, 4);   // 4 waves of 32 per block

    // degrees / symmetric norm (edge-topology only; shared by all layers)
    fill_deg<<<cdiv(N_NODES, B), B, 0, stream>>>(dinv, N_NODES);
    count_deg<<<cdiv(N_EDGES, B), B, 0, stream>>>(dst, dinv, N_EDGES);
    rsqrt_deg<<<cdiv(N_NODES, B), B, 0, stream>>>(dinv, N_NODES);

    // convert weights (transposed, f16) and input activations
    convert_w_t<<<cdiv(DIM_H * DIM_H, B), B, 0, stream>>>(W0, Wt0, DIM_H, DIM_H, DIM_H);
    convert_w_t<<<cdiv(DIM_H * DIM_H, B), B, 0, stream>>>(W1, Wt1, DIM_H, DIM_H, DIM_H);
    convert_w_t<<<cdiv(DIM_OUT_PAD * DIM_H, B), B, 0, stream>>>(W2, Wt2, DIM_H, DIM_OUT, DIM_OUT_PAD);
    convert_x_f16<<<cdiv((long long)N_NODES * DIM_H, B), B, 0, stream>>>(x, xh, N_NODES * DIM_H);

    // ---- layer 1 ----
    gemm_wmma_f16<<<gemmBlocks, 128, 0, stream>>>(xh, Wt0, h, nRowTiles, DIM_H / 16, DIM_H, DIM_H);
    self_init<<<cdiv((long long)N_NODES * DIM_H, B), B, 0, stream>>>(h, dinv, b0, agg, N_NODES, DIM_H, DIM_H);
    edge_scatter<<<cdiv((long long)N_EDGES * (DIM_H / 4), B), B, 0, stream>>>(src, dst, h, dinv, agg, N_EDGES, DIM_H / 4, DIM_H);
    relu_to_f16<<<cdiv((long long)N_NODES * DIM_H, B), B, 0, stream>>>(agg, xh, N_NODES * DIM_H);

    // ---- layer 2 ----
    gemm_wmma_f16<<<gemmBlocks, 128, 0, stream>>>(xh, Wt1, h, nRowTiles, DIM_H / 16, DIM_H, DIM_H);
    self_init<<<cdiv((long long)N_NODES * DIM_H, B), B, 0, stream>>>(h, dinv, b1, agg, N_NODES, DIM_H, DIM_H);
    edge_scatter<<<cdiv((long long)N_EDGES * (DIM_H / 4), B), B, 0, stream>>>(src, dst, h, dinv, agg, N_EDGES, DIM_H / 4, DIM_H);
    relu_to_f16<<<cdiv((long long)N_NODES * DIM_H, B), B, 0, stream>>>(agg, xh, N_NODES * DIM_H);

    // ---- layer 3 (OUT=40 padded to 48) ----
    gemm_wmma_f16<<<gemmBlocks, 128, 0, stream>>>(xh, Wt2, h, nRowTiles, DIM_OUT_PAD / 16, DIM_H, DIM_OUT_PAD);
    self_init<<<cdiv((long long)N_NODES * DIM_OUT_PAD, B), B, 0, stream>>>(h, dinv, b2, agg, N_NODES, DIM_OUT_PAD, DIM_OUT);
    edge_scatter<<<cdiv((long long)N_EDGES * (DIM_OUT / 4), B), B, 0, stream>>>(src, dst, h, dinv, agg, N_EDGES, DIM_OUT / 4, DIM_OUT_PAD);
    relu_logsoftmax40<<<cdiv(N_NODES, 8), 256, 0, stream>>>(agg, out, N_NODES, DIM_OUT_PAD);
}
